// CosineSimilarity2D_9663676416670
// MI455X (gfx1250) — matrix operand order repaired and checked by reading, not compiled, not run on hardware
//
#include <hip/hip_runtime.h>
#include <hip/hip_bf16.h>

typedef _Float16 half8  __attribute__((ext_vector_type(8)));
typedef _Float16 half16 __attribute__((ext_vector_type(16)));
typedef float    f32x8  __attribute__((ext_vector_type(8)));
typedef int      v4i    __attribute__((vector_size(16)));

#define N_IMG 16
#define C_IN  64
#define HW    128
#define O_CH  64
#define KSZ   16
#define HO    113
#define PLANE (HO*HO)          // 12769
#define MTOT  (N_IMG*PLANE)    // 204304
#define KTOT  (C_IN*KSZ*KSZ)   // 16384

#define BM 128
#define BN 64
#define APITCH 72
#define BPITCH 72
#define ABUF (BM*APITCH)       // halfs per A buffer
#define BBUF (BN*BPITCH)       // halfs per B buffer

// ---------------------------------------------------------------------------
// Async global->LDS copy of 16 bytes (CDNA5 ASYNCcnt path). No VGPR data path.
// ---------------------------------------------------------------------------
__device__ __forceinline__ void async_copy_b128(const _Float16* g, _Float16* l) {
#if __has_builtin(__builtin_amdgcn_global_load_async_to_lds_b128)
  void* gv = (void*)g;  // drop const before the address-space cast
  void* lv = (void*)l;
  __builtin_amdgcn_global_load_async_to_lds_b128(
      (__attribute__((address_space(1))) v4i*)gv,
      (__attribute__((address_space(3))) v4i*)lv, 0, 0);
#else
  unsigned lofs = (unsigned)(size_t)(__attribute__((address_space(3))) void*)(void*)l;
  asm volatile("global_load_async_to_lds_b128 %0, %1, off"
               :: "v"(lofs), "v"(g) : "memory");
#endif
}

__device__ __forceinline__ void wait_async0() {
#if __has_builtin(__builtin_amdgcn_s_wait_asynccnt)
  __builtin_amdgcn_s_wait_asynccnt(0);
#else
  asm volatile("s_wait_asynccnt 0x0" ::: "memory");
#endif
}

// ---------------------------------------------------------------------------
// Kernel 1: x NCHW f32 -> NHWC f16, plus per-pixel channel sum-of-squares.
// ---------------------------------------------------------------------------
__global__ __launch_bounds__(256)
void prep_x_kernel(const float* __restrict__ x,
                   _Float16* __restrict__ xh,
                   float* __restrict__ colsq) {
  int p = blockIdx.x * blockDim.x + threadIdx.x;   // pixel index (n,h,w)
  if (p >= N_IMG * HW * HW) return;
  int n  = p >> 14;            // /(128*128)
  int hw = p & (HW * HW - 1);
  const float* xp = x + ((size_t)n * C_IN << 14) + hw;
  float ss = 0.f;
  _Float16* dst = xh + (size_t)p * C_IN;
  #pragma unroll
  for (int cb = 0; cb < C_IN / 8; ++cb) {
    half8 h;
    #pragma unroll
    for (int c8 = 0; c8 < 8; ++c8) {
      float v = xp[(size_t)(cb * 8 + c8) << 14];
      ss += v * v;
      h[c8] = (_Float16)v;
    }
    *(half8*)(dst + cb * 8) = h;   // 16B aligned store
  }
  colsq[p] = ss;
}

// ---------------------------------------------------------------------------
// Kernel 2: per-output-window inverse L2 norm of x (16x16 box over colsq).
// ---------------------------------------------------------------------------
__global__ __launch_bounds__(256)
void xnorm_kernel(const float* __restrict__ colsq,
                  float* __restrict__ xinv) {
  int m = blockIdx.x * blockDim.x + threadIdx.x;
  if (m >= MTOT) return;
  int n = m / PLANE; int rem = m - n * PLANE;
  int oh = rem / HO; int ow = rem - oh * HO;
  const float* base = colsq + ((size_t)n * HW * HW) + oh * HW + ow;
  float s = 0.f;
  for (int kh = 0; kh < KSZ; ++kh) {
    float r = 0.f;
    #pragma unroll
    for (int kw = 0; kw < KSZ; ++kw) r += base[kh * HW + kw];
    s += r;
  }
  xinv[m] = rsqrtf(s);
}

// ---------------------------------------------------------------------------
// Kernel 3: normalize each filter of y to unit L2, store f16 as [kh][kw][o][c].
// ---------------------------------------------------------------------------
__global__ __launch_bounds__(256)
void prep_y_kernel(const float* __restrict__ y,
                   _Float16* __restrict__ Bmat) {
  int o = blockIdx.x;
  int t = threadIdx.x;
  const float* yo = y + (size_t)o * KTOT;
  float s = 0.f;
  for (int e = t; e < KTOT; e += 256) { float v = yo[e]; s += v * v; }
  __shared__ float red[256];
  red[t] = s; __syncthreads();
  for (int w = 128; w > 0; w >>= 1) {
    if (t < w) red[t] += red[t + w];
    __syncthreads();
  }
  float inv = rsqrtf(red[0]);
  for (int e = t; e < KTOT; e += 256) {
    int c  = e >> 8;          // y layout OIHW: e = c*256 + kh*16 + kw
    int kh = (e >> 4) & 15;
    int kw = e & 15;
    Bmat[(size_t)(((kh * KSZ + kw) * O_CH + o) * C_IN + c)] =
        (_Float16)(yo[e] * inv);
  }
}

// ---------------------------------------------------------------------------
// Kernel 4: implicit-GEMM conv via v_wmma_f32_16x16x32_f16.
// Block tile: 128(M) x 64(N); 8 waves, each wave a 32x32 C tile.
// 256 K-stages (one (kh,kw) tap, BK=64 channels), double-buffered LDS filled
// by async global->LDS copies overlapped with WMMA compute.
// ---------------------------------------------------------------------------
__global__ __launch_bounds__(256)
void ncc_gemm_kernel(const _Float16* __restrict__ xh,
                     const _Float16* __restrict__ Bmat,
                     const float* __restrict__ xinv,
                     float* __restrict__ out) {
  __shared__ __align__(16) _Float16 As[2 * ABUF];
  __shared__ __align__(16) _Float16 Bs[2 * BBUF];

  const int t    = threadIdx.x;
  const int m0   = blockIdx.x * BM;
  const int lane = t & 31;
  const int wv   = t >> 5;
  const int R0   = (wv & 3) * 32;   // M offset of this wave's tile
  const int O0   = (wv >> 2) * 32;  // N offset of this wave's tile
  const int l16  = lane & 15;
  const int lh   = lane >> 4;       // lane half (0: lanes 0-15, 1: 16-31)

  // -------- staging addresses; all int divisions hoisted out of the loop ----
  const int rq = t >> 3;            // 0..31
  const int q  = t & 7;             // 8-half chunk within a 64-half row
  const _Float16* aG[4];            // per-thread global src (tap 0)
  _Float16*       aL[4];            // per-thread LDS dst (buffer 0)
  #pragma unroll
  for (int i = 0; i < 4; ++i) {
    int r = rq + 32 * i;                           // A-tile row 0..127
    int m = m0 + r; if (m >= MTOT) m = MTOT - 1;   // clamp tail reads
    int n = m / PLANE; int rem = m - n * PLANE;
    int oh = rem / HO; int ow = rem - oh * HO;
    aG[i] = xh + ((n * HW + oh) * HW + ow) * C_IN + q * 8;
    aL[i] = &As[r * APITCH + q * 8];
  }
  const _Float16* bG[2];
  _Float16*       bL[2];
  #pragma unroll
  for (int i = 0; i < 2; ++i) {
    int o = rq + 32 * i;                           // B-tile row (out channel)
    bG[i] = Bmat + o * C_IN + q * 8;
    bL[i] = &Bs[o * BPITCH + q * 8];
  }

  f32x8 acc[2][2] = {};

  const int NSTAGE = KSZ * KSZ;     // 256 taps

  // prologue: stage 0 -> buffer 0
  #pragma unroll
  for (int i = 0; i < 4; ++i) async_copy_b128(aG[i], aL[i]);
  #pragma unroll
  for (int i = 0; i < 2; ++i) async_copy_b128(bG[i], bL[i]);

  for (int s = 0; s < NSTAGE; ++s) {
    const int buf = s & 1;
    wait_async0();        // this wave's copies for stage s are in LDS
    __syncthreads();      // ...and so are everyone else's; buf^1 readers done

    if (s + 1 < NSTAGE) { // issue stage s+1 into the other buffer (overlaps)
      const int sn = s + 1;
      const int kh = sn >> 4, kw = sn & 15;
      const int aoff = (kh * HW + kw) * C_IN;      // tap shift in NHWC
      const int boff = sn * (O_CH * C_IN);
      const int lbuf = (buf ^ 1);
      #pragma unroll
      for (int i = 0; i < 4; ++i)
        async_copy_b128(aG[i] + aoff, aL[i] + lbuf * ABUF);
      #pragma unroll
      for (int i = 0; i < 2; ++i)
        async_copy_b128(bG[i] + boff, bL[i] + lbuf * BBUF);
    }

    // ---- compute stage s from LDS buffer `buf` ----
    const _Float16* Ab = &As[buf * ABUF];
    const _Float16* Bb = &Bs[buf * BBUF];
    #pragma unroll
    for (int ks = 0; ks < 2; ++ks) {      // two K=32 sub-steps of BK=64
      half16 a[2], b[2];
      // A fragment (16x32 f16): lanes 0-15 hold K {0..7,16..23},
      // lanes 16-31 hold K {8..15,24..31} of this K=32 slice.
      #pragma unroll
      for (int mi = 0; mi < 2; ++mi) {
        const _Float16* p = Ab + (R0 + mi * 16 + l16) * APITCH + ks * 32 + lh * 8;
        half8 lo = *(const half8*)p;
        half8 hi = *(const half8*)(p + 16);
        a[mi] = __builtin_shufflevector(lo, hi,
                 0,1,2,3,4,5,6,7,8,9,10,11,12,13,14,15);
      }
      // B fragment (32x16 f16): lane = column; lanes 0-15 K {0..7,16..23},
      // lanes 16-31 K {8..15,24..31} (same order as A).
      #pragma unroll
      for (int ni = 0; ni < 2; ++ni) {
        const _Float16* p = Bb + (O0 + ni * 16 + l16) * BPITCH + ks * 32 + lh * 8;
        half8 lo = *(const half8*)p;
        half8 hi = *(const half8*)(p + 16);
        b[ni] = __builtin_shufflevector(lo, hi,
                 0,1,2,3,4,5,6,7,8,9,10,11,12,13,14,15);
      }
      #pragma unroll
      for (int mi = 0; mi < 2; ++mi)
        #pragma unroll
        for (int ni = 0; ni < 2; ++ni)
          acc[mi][ni] = __builtin_amdgcn_wmma_f32_16x16x32_f16(
              false, a[mi], false, b[ni], (short)0, acc[mi][ni],
              false, false);
    }
  }

  // -------- fused epilogue: scale by window inv-norm, ReLU, NCHW store -----
  #pragma unroll
  for (int mi = 0; mi < 2; ++mi) {
    #pragma unroll
    for (int ni = 0; ni < 2; ++ni) {
      int o = O0 + ni * 16 + l16;             // N = lane%16 in C/D layout
      #pragma unroll
      for (int r = 0; r < 8; ++r) {
        int M = R0 + mi * 16 + lh * 8 + r;    // VGPR r: M=r / M=r+8 per half
        int m = m0 + M;
        if (m < MTOT) {
          int n = m / PLANE; int rem = m - n * PLANE;
          float v = acc[mi][ni][r] * xinv[m];
          v = v > 0.f ? v : 0.f;
          out[(size_t)(n * O_CH + o) * PLANE + rem] = v;
        }
      }
    }
  }
}

// ---------------------------------------------------------------------------
extern "C" void kernel_launch(void* const* d_in, const int* in_sizes, int n_in,
                              void* d_out, int out_size, void* d_ws, size_t ws_size,
                              hipStream_t stream) {
  (void)in_sizes; (void)n_in; (void)out_size; (void)ws_size;
  const float* x = (const float*)d_in[0];
  const float* y = (const float*)d_in[1];
  float* out = (float*)d_out;

  char* ws = (char*)d_ws;
  size_t off = 0;
  auto carve = [&](size_t bytes) -> void* {
    void* p = ws + off;
    off += (bytes + 255) & ~(size_t)255;
    return p;
  };
  _Float16* xh   = (_Float16*)carve((size_t)N_IMG * HW * HW * C_IN * 2); // 32 MB
  _Float16* Bmat = (_Float16*)carve((size_t)KTOT * O_CH * 2);            //  2 MB
  float*    colsq= (float*)carve((size_t)N_IMG * HW * HW * 4);           //  1 MB
  float*    xinv = (float*)carve((size_t)MTOT * 4);                      // .8 MB

  hipLaunchKernelGGL(prep_x_kernel, dim3((N_IMG*HW*HW + 255) / 256), dim3(256),
                     0, stream, x, xh, colsq);
  hipLaunchKernelGGL(xnorm_kernel, dim3((MTOT + 255) / 256), dim3(256),
                     0, stream, colsq, xinv);
  hipLaunchKernelGGL(prep_y_kernel, dim3(O_CH), dim3(256),
                     0, stream, y, Bmat);
  hipLaunchKernelGGL(ncc_gemm_kernel, dim3((MTOT + BM - 1) / BM), dim3(256),
                     0, stream, xh, Bmat, xinv, out);
}
